// AdeptPolylineEncoder_54408645705944
// MI455X (gfx1250) — matrix operand order repaired and checked by reading, not compiled, not run on hardware
//
#include <hip/hip_runtime.h>
#include <cfloat>

typedef __attribute__((ext_vector_type(16))) _Float16 v16h;
typedef __attribute__((ext_vector_type(8)))  _Float16 v8h;
typedef __attribute__((ext_vector_type(8)))  float    v8f;

// ---- Problem dims ----
#define BB 4
#define TT 128
#define PP 1024
#define CC 32
#define NBT (BB*TT)          // 512 (b,t) pairs

// ---- WMMA fragment index helpers (CDNA5 ISA 7.12.2, wave32) ----
// A 16x32 f16: lane L holds row M=L&15; element e -> K = 16*(e>=8) + (e&7) + 8*(L>=16)
__device__ __forceinline__ int a_kidx(int lane, int e) {
  return ((e & 8) << 1) + (e & 7) + ((lane >> 4) << 3);
}
// B 32x16 f16: lane L holds col N=L&15; element e -> K = 16*(e>=8) + (e&7) + 8*(L>=16)
__device__ __forceinline__ int b_kidx(int lane, int e) {
  return ((e & 8) << 1) + (e & 7) + ((lane >> 4) << 3);
}
// C/D 16x16 f32: VGPR r, lane L -> row = r + 8*(L>=16), col = L&15

__device__ __forceinline__ v8f wmma16(v16h a, v16h b, v8f c) {
  return __builtin_amdgcn_wmma_f32_16x16x32_f16(false, a, false, b, (short)0, c, false, false);
}

// single-instruction ReLU: v_med3_f32(v, 0, +FLT_MAX) == clamp(v, 0, max)
__device__ __forceinline__ float relu(float v) {
  return __builtin_amdgcn_fmed3f(v, 0.0f, FLT_MAX);
}

__device__ __forceinline__ void atomicMaxNonNeg(float* addr, float v) {
  atomicMax((int*)addr, __float_as_int(v));   // valid: relu'd values are >= 0
}

// A-fragment assembly from a 32-half-wide k-step row slice in LDS/global (f16).
// Pattern = two contiguous 8-half (16B) runs: [hi, hi+8) and [16+hi, 24+hi).
__device__ __forceinline__ v16h load_a_frag_h(const _Float16* base, int lane) {
  const v8h* p = (const v8h*)base;
  const int q = lane >> 4;
  v8h lo = p[q], hi = p[2 + q];
  v16h t;
#pragma unroll
  for (int e = 0; e < 8; ++e) { t[e] = lo[e]; t[e + 8] = hi[e]; }
  return t;
}
// Same but from f32 memory (two 32B v8f runs), converting to f16.
__device__ __forceinline__ v16h load_a_frag_f32(const float* base, int lane) {
  const v8f* p = (const v8f*)base;
  const int q = lane >> 4;
  v8f lo = p[q], hi = p[2 + q];
  v16h t;
#pragma unroll
  for (int e = 0; e < 8; ++e) { t[e] = (_Float16)lo[e]; t[e + 8] = (_Float16)hi[e]; }
  return t;
}

// =====================================================================
// Kernel 0: pack all weights to f16 in B-fragment register order.
// dst[((nt*kstotal + ks)*32 + lane)] = v16h fragment for (nt, ks).
// grid = 532 blocks x 32 threads (one tile per block).
// =====================================================================
__device__ __forceinline__ void pack_tile(const float* __restrict__ W, int ncol,
                                          int nt, int ks, v16h* __restrict__ dst,
                                          int kstotal) {
  const int lane = threadIdx.x & 31;
  const int col  = lane & 15;
  v16h t;
#pragma unroll
  for (int e = 0; e < 16; ++e)
    t[e] = (_Float16)W[(ks * 32 + b_kidx(lane, e)) * ncol + nt * 16 + col];
  dst[(nt * kstotal + ks) * 32 + lane] = t;
}

__global__ void __launch_bounds__(32)
k_pack(const float* __restrict__ W1, const float* __restrict__ W2,
       const float* __restrict__ W3, const float* __restrict__ W4,
       const float* __restrict__ Wo1, const float* __restrict__ Wo2,
       v16h* __restrict__ W1p, v16h* __restrict__ W2p,
       v16h* __restrict__ W3p, v16h* __restrict__ W4p,
       v16h* __restrict__ Wo1p, v16h* __restrict__ Wo2p) {
  int t = blockIdx.x;
  if      (t < 4)   {            pack_tile(W1,  64,  t,      0,     W1p,  1); }
  else if (t < 20)  { t -= 4;    pack_tile(W2,  128, t >> 1, t & 1, W2p,  2); }
  else if (t < 148) { t -= 20;   pack_tile(W3,  256, t >> 3, t & 7, W3p,  8); }
  else if (t < 276) { t -= 148;  pack_tile(W4,  256, t >> 3, t & 7, W4p,  8); }
  else if (t < 404) { t -= 276;  pack_tile(Wo1, 256, t >> 3, t & 7, Wo1p, 8); }
  else              { t -= 404;  pack_tile(Wo2, 256, t >> 3, t & 7, Wo2p, 8); }
}

// =====================================================================
// Kernel 1: pre-MLP 32->64->128 (+ReLU, mask-zero); H2 written in packed
// A-fragment order; pooled max; valid flag.
// grid = 512 (one block per (b,t)), block = 256 threads = 8 waves
// =====================================================================
__global__ void __launch_bounds__(256)
k_pre(const float* __restrict__ X, const unsigned char* __restrict__ mask,
      const v16h* __restrict__ W1p, const float* __restrict__ b1,
      const v16h* __restrict__ W2p, const float* __restrict__ b2,
      v16h* __restrict__ H2p, float* __restrict__ pooled,
      int* __restrict__ validArr) {
  __shared__ _Float16 sHs[8][16][64];    // 16 KB: GEMM1 -> GEMM2 bridge (wave-private)
  __shared__ _Float16 sH2r[8][16][128];  // 32 KB: D-layout -> A-layout repack (wave-private)
  __shared__ float    pmax[128];
  __shared__ int      anyv;

  const int bt   = blockIdx.x;
  const int tid  = threadIdx.x;
  const int lane = tid & 31;
  const int w    = tid >> 5;
  const int col  = lane & 15;
  const int hi8  = (lane >> 4) << 3;

  if (tid < 128) pmax[tid] = 0.f;
  if (tid == 0)  anyv = 0;
  __syncthreads();

  int my_any = 0;
  for (int p = tid; p < PP; p += 256) my_any |= (mask[bt * PP + p] ? 1 : 0);
  if (my_any) atomicOr(&anyv, 1);

  float regmax[8];
#pragma unroll
  for (int i = 0; i < 8; ++i) regmax[i] = 0.f;

  for (int mt = w; mt < PP / 16; mt += 8) {   // 8 M-tiles per wave, uniform trip count
    const int rowbase = mt * 16;

    // ---- GEMM1: A (16x32) from global points (two v8f runs per lane) ----
    const float* xrow = X + ((long)bt * PP + rowbase + (lane & 15)) * CC;
    v16h a1 = load_a_frag_f32(xrow, lane);

    // local mask bytes for this tile's rows
    unsigned char mloc[8];
#pragma unroll
    for (int r = 0; r < 8; ++r) mloc[r] = mask[bt * PP + rowbase + hi8 + r];

#pragma unroll 2
    for (int nt = 0; nt < 4; ++nt) {          // N = 64
      const float bias = b1[nt * 16 + col];
      v8f acc = {bias, bias, bias, bias, bias, bias, bias, bias};
      acc = wmma16(a1, W1p[nt * 32 + lane], acc);
#pragma unroll
      for (int r = 0; r < 8; ++r)
        sHs[w][r + hi8][nt * 16 + col] = (_Float16)relu(acc[r]);
    }
    __syncthreads();

    // ---- GEMM2: (16x64)@(64x128), 2 k-steps ----
    v16h a2[2];
#pragma unroll
    for (int ks = 0; ks < 2; ++ks)
      a2[ks] = load_a_frag_h(&sHs[w][lane & 15][ks * 32], lane);

#pragma unroll 2
    for (int nt = 0; nt < 8; ++nt) {          // N = 128
      const float bias = b2[nt * 16 + col];
      v8f acc = {bias, bias, bias, bias, bias, bias, bias, bias};
#pragma unroll
      for (int ks = 0; ks < 2; ++ks)
        acc = wmma16(a2[ks], W2p[(nt * 2 + ks) * 32 + lane], acc);
      float rm = regmax[nt];
#pragma unroll
      for (int r = 0; r < 8; ++r) {
        float v = relu(acc[r]);               // ReLU
        if (!mloc[r]) v = 0.f;                // mask-zero
        sH2r[w][r + hi8][nt * 16 + col] = (_Float16)v;
        rm = fmaxf(rm, v);
      }
      regmax[nt] = rm;
    }
    __syncthreads();

    // ---- repack D-layout tile -> packed A-fragment layout in global H2p ----
#pragma unroll
    for (int ks = 0; ks < 4; ++ks) {          // 128 cols = 4 k-steps
      v16h t = load_a_frag_h(&sH2r[w][lane & 15][ks * 32], lane);
      H2p[(((long)bt * 64 + mt) * 4 + ks) * 32 + lane] = t;
    }
    __syncthreads();
  }

  // merge pooled maxima (one atomic per owned column strip)
#pragma unroll
  for (int nt = 0; nt < 8; ++nt) atomicMaxNonNeg(&pmax[nt * 16 + col], regmax[nt]);
  __syncthreads();
  if (tid < 128) pooled[bt * 128 + tid] = pmax[tid];
  if (tid == 0)  validArr[bt] = anyv;
}

// =====================================================================
// Kernel 2: [H2 | pooled] (256) -> W3 -> ReLU -> W4 -> ReLU -> mask -> max over P
// grid = 512, block = 256 (8 waves).
// Broadcast trick: pooled@W3[128:256]+b3 is row-independent -> computed once per
// block into poolc[256]; per-row GEMM1 then runs with K=128 only.
// Wave w owns an M-tile *pair* (wm2 = w&1) and an N-quarter (ntlo = (w>>1)*4),
// so each B-fragment load feeds 2 WMMAs.
// =====================================================================
__global__ void __launch_bounds__(256)
k_mid(const v16h* __restrict__ H2p, const float* __restrict__ pooled,
      const unsigned char* __restrict__ mask,
      const v16h* __restrict__ W3p, const float* __restrict__ b3,
      const v16h* __restrict__ W4p, const float* __restrict__ b4,
      float* __restrict__ feat) {
  __shared__ _Float16 sHmid[64][256];   // 32 KB intermediate chunk
  __shared__ float    fmax[256];
  __shared__ float    poolc[256];       // pooled @ W3[128:256] + b3 (row-broadcast)

  const int bt   = blockIdx.x;
  const int tid  = threadIdx.x;
  const int lane = tid & 31;
  const int w    = tid >> 5;
  const int col  = lane & 15;
  const int hi8  = (lane >> 4) << 3;
  const int wm2  = w & 1;               // which M-tile pair within the 64-row chunk
  const int ntlo = (w >> 1) * 4;        // N-tile quarter [ntlo, ntlo+4)

  __builtin_prefetch(W3p, 0, 0);        // global_prefetch_b8 into caches
  __builtin_prefetch(W4p, 0, 0);

  fmax[tid] = 0.f;

  // ---- phase 0: poolc = pooled @ W3[128:256] + b3 (once per block) ----
  {
    v16h a_pool[4];                     // every A row = pooled vector (broadcast)
#pragma unroll
    for (int ks = 0; ks < 4; ++ks)
      a_pool[ks] = load_a_frag_f32(pooled + bt * 128 + ks * 32, lane);
#pragma unroll
    for (int i = 0; i < 2; ++i) {       // wave w computes N-tiles 2w, 2w+1
      const int nt = w * 2 + i;
      const float bias = b3[nt * 16 + col];
      v8f acc = {bias, bias, bias, bias, bias, bias, bias, bias};
#pragma unroll
      for (int ks = 0; ks < 4; ++ks)
        acc = wmma16(a_pool[ks], W3p[(nt * 8 + 4 + ks) * 32 + lane], acc);
      if (lane < 16) poolc[nt * 16 + col] = acc[0];   // all D rows identical
    }
  }

  float regmax[4];
#pragma unroll
  for (int i = 0; i < 4; ++i) regmax[i] = 0.f;
  __syncthreads();

  for (int chunk = 0; chunk < 16; ++chunk) {
    const int mtb = chunk * 4 + wm2 * 2;        // first M-tile of this wave's pair

    // A fragments (H2 half only, K=128): packed v16h loads, 2 M-tiles
    v16h aH[2][4];
#pragma unroll
    for (int m = 0; m < 2; ++m)
#pragma unroll
      for (int ks = 0; ks < 4; ++ks)
        aH[m][ks] = H2p[(((long)bt * 64 + mtb + m) * 4 + ks) * 32 + lane];

    unsigned char mloc[2][8];
#pragma unroll
    for (int m = 0; m < 2; ++m)
#pragma unroll
      for (int r = 0; r < 8; ++r)
        mloc[m][r] = mask[bt * PP + (mtb + m) * 16 + hi8 + r];

#pragma unroll 1
    for (int nti = 0; nti < 4; ++nti) {         // keep live ranges small
      const int nt = ntlo + nti;
      const float base = poolc[nt * 16 + col];  // bias + pooled contribution
      v8f acc0 = {base, base, base, base, base, base, base, base};
      v8f acc1 = acc0;
#pragma unroll
      for (int ks = 0; ks < 4; ++ks) {
        const v16h bf = W3p[(nt * 8 + ks) * 32 + lane];
        acc0 = wmma16(aH[0][ks], bf, acc0);
        acc1 = wmma16(aH[1][ks], bf, acc1);
      }
#pragma unroll
      for (int r = 0; r < 8; ++r) {
        sHmid[(wm2 * 2 + 0) * 16 + r + hi8][nt * 16 + col] = (_Float16)relu(acc0[r]);
        sHmid[(wm2 * 2 + 1) * 16 + r + hi8][nt * 16 + col] = (_Float16)relu(acc1[r]);
      }
    }
    __syncthreads();   // other waves fill the remaining columns of our rows

    // ---- GEMM2: full K=256, 2 M-tiles per wave ----
    v16h a2[2][8];
#pragma unroll
    for (int m = 0; m < 2; ++m)
#pragma unroll
      for (int ks = 0; ks < 8; ++ks)
        a2[m][ks] = load_a_frag_h(&sHmid[(wm2 * 2 + m) * 16 + (lane & 15)][ks * 32], lane);

#pragma unroll 1
    for (int nti = 0; nti < 4; ++nti) {
      const int nt = ntlo + nti;
      const float bias = b4[nt * 16 + col];
      v8f acc0 = {bias, bias, bias, bias, bias, bias, bias, bias};
      v8f acc1 = acc0;
#pragma unroll
      for (int ks = 0; ks < 8; ++ks) {
        const v16h bf = W4p[(nt * 8 + ks) * 32 + lane];
        acc0 = wmma16(a2[0][ks], bf, acc0);
        acc1 = wmma16(a2[1][ks], bf, acc1);
      }
      float rm = regmax[nti];
#pragma unroll
      for (int r = 0; r < 8; ++r) {
        float v0 = relu(acc0[r]);
        float v1 = relu(acc1[r]);
        if (!mloc[0][r]) v0 = 0.f;
        if (!mloc[1][r]) v1 = 0.f;
        rm = fmaxf(rm, fmaxf(v0, v1));
      }
      regmax[nti] = rm;
    }
    __syncthreads();   // protect sHmid before next chunk overwrites
  }

#pragma unroll
  for (int nti = 0; nti < 4; ++nti)
    atomicMaxNonNeg(&fmax[(ntlo + nti) * 16 + col], regmax[nti]);
  __syncthreads();
  feat[bt * 256 + tid] = fmax[tid];
}

// =====================================================================
// Kernel 3: out MLP on 512 rows: relu(feat@Wo1+bo1)@Wo2+bo2, zero invalid rows
// grid = 8 blocks x 128 threads (4 waves); wave handles one 16-row M-tile
// =====================================================================
__global__ void __launch_bounds__(128)
k_out(const float* __restrict__ feat, const int* __restrict__ validArr,
      const v16h* __restrict__ Wo1p, const float* __restrict__ bo1,
      const v16h* __restrict__ Wo2p, const float* __restrict__ bo2,
      float* __restrict__ out) {
  __shared__ _Float16 sH[4][16][256];   // 32 KB, per-wave bridge

  const int tid  = threadIdx.x;
  const int lane = tid & 31;
  const int w    = tid >> 5;
  const int mt   = blockIdx.x * 4 + w;  // 32 M-tiles total
  const int rowbase = mt * 16;
  const int col  = lane & 15;
  const int hi8  = (lane >> 4) << 3;

  // GEMM1: A from feat (f32, two v8f runs per k-step)
  v16h a[8];
#pragma unroll
  for (int ks = 0; ks < 8; ++ks)
    a[ks] = load_a_frag_f32(feat + (rowbase + (lane & 15)) * 256 + ks * 32, lane);

#pragma unroll 1
  for (int nt = 0; nt < 16; ++nt) {
    const float bias = bo1[nt * 16 + col];
    v8f acc = {bias, bias, bias, bias, bias, bias, bias, bias};
#pragma unroll
    for (int ks = 0; ks < 8; ++ks)
      acc = wmma16(a[ks], Wo1p[(nt * 8 + ks) * 32 + lane], acc);
#pragma unroll
    for (int r = 0; r < 8; ++r)
      sH[w][r + hi8][nt * 16 + col] = (_Float16)relu(acc[r]);   // ReLU
  }
  __syncthreads();

  // GEMM2 (no final ReLU), then zero rows with no valid points
  v16h a2[8];
#pragma unroll
  for (int ks = 0; ks < 8; ++ks)
    a2[ks] = load_a_frag_h(&sH[w][lane & 15][ks * 32], lane);

#pragma unroll 1
  for (int nt = 0; nt < 16; ++nt) {
    const float bias = bo2[nt * 16 + col];
    v8f acc = {bias, bias, bias, bias, bias, bias, bias, bias};
#pragma unroll
    for (int ks = 0; ks < 8; ++ks)
      acc = wmma16(a2[ks], Wo2p[(nt * 8 + ks) * 32 + lane], acc);
#pragma unroll
    for (int r = 0; r < 8; ++r) {
      const int row = rowbase + r + hi8;   // row index == flattened (b,t)
      float v = acc[r];
      if (!validArr[row]) v = 0.f;
      out[row * 256 + nt * 16 + col] = v;
    }
  }
}

// =====================================================================
extern "C" void kernel_launch(void* const* d_in, const int* in_sizes, int n_in,
                              void* d_out, int out_size, void* d_ws, size_t ws_size,
                              hipStream_t stream) {
  const float*         X    = (const float*)d_in[0];
  const unsigned char* mask = (const unsigned char*)d_in[1];   // jax bool -> 1 byte
  const float* W1 = (const float*)d_in[2];  const float* b1 = (const float*)d_in[3];
  const float* W2 = (const float*)d_in[4];  const float* b2 = (const float*)d_in[5];
  const float* W3 = (const float*)d_in[6];  const float* b3 = (const float*)d_in[7];
  const float* W4 = (const float*)d_in[8];  const float* b4 = (const float*)d_in[9];
  const float* Wo1 = (const float*)d_in[10]; const float* bo1 = (const float*)d_in[11];
  const float* Wo2 = (const float*)d_in[12]; const float* bo2 = (const float*)d_in[13];

  // ---- workspace layout (all offsets 64B-aligned) ----
  char* ws = (char*)d_ws;
  size_t off = 0;
  v16h* H2p = (v16h*)(ws + off);  off += (size_t)NBT * 64 * 4 * 32 * sizeof(v16h); // 128 MB
  float* pooled = (float*)(ws + off);  off += (size_t)NBT * 128 * sizeof(float);
  float* feat   = (float*)(ws + off);  off += (size_t)NBT * 256 * sizeof(float);
  int*   valid  = (int*)(ws + off);    off += (size_t)NBT * sizeof(int);
  v16h* W1p  = (v16h*)(ws + off);  off += (size_t)4   * 1 * 32 * sizeof(v16h);
  v16h* W2p  = (v16h*)(ws + off);  off += (size_t)8   * 2 * 32 * sizeof(v16h);
  v16h* W3p  = (v16h*)(ws + off);  off += (size_t)16  * 8 * 32 * sizeof(v16h);
  v16h* W4p  = (v16h*)(ws + off);  off += (size_t)16  * 8 * 32 * sizeof(v16h);
  v16h* Wo1p = (v16h*)(ws + off);  off += (size_t)16  * 8 * 32 * sizeof(v16h);
  v16h* Wo2p = (v16h*)(ws + off);  off += (size_t)16  * 8 * 32 * sizeof(v16h);

  k_pack<<<532, 32, 0, stream>>>(W1, W2, W3, W4, Wo1, Wo2,
                                 W1p, W2p, W3p, W4p, Wo1p, Wo2p);
  k_pre<<<NBT, 256, 0, stream>>>(X, mask, W1p, b1, W2p, b2, H2p, pooled, valid);
  k_mid<<<NBT, 256, 0, stream>>>(H2p, pooled, mask, W3p, b3, W4p, b4, feat);
  k_out<<<8, 128, 0, stream>>>(feat, valid, Wo1p, bo1, Wo2p, bo2, (float*)d_out);
}
